// MMINet_26405458935977
// MI455X (gfx1250) — matrix operand order repaired and checked by reading, not compiled, not run on hardware
//
#include <hip/hip_runtime.h>
#include <hip/hip_bf16.h>

// ---------------------------------------------------------------------------
// MMINet (Conv-TasNet style) for gfx1250. All channel matmuls run on
// v_wmma_f32_16x16x32_bf16 (f32 accumulate). Activations are channel-last
// bf16 so both WMMA fragments are contiguous global_load_b128. Each wave
// computes a 16M x (16*NT)N strip, reusing its A fragment across NT B tiles,
// with software-pipelined (double-buffered) fragment loads so each k-step's
// WMMAs consume registers loaded one full step earlier.
// ---------------------------------------------------------------------------

typedef __attribute__((ext_vector_type(16))) __bf16 v16bf;
typedef __attribute__((ext_vector_type(8)))  __bf16 v8bf;
typedef __attribute__((ext_vector_type(8)))  float  v8f;

#define LF    4095            // frames per batch: (32768-16)/8 + 1
#define MT    (4 * LF)        // total positions (batch*frames) = 16380
#define SEQN  32768
#define CH    256
#define EPSLN 1e-5f

__device__ __forceinline__ float  bf2f(__bf16 x) { return (float)x; }
__device__ __forceinline__ __bf16 f2bf(float  x) { return (__bf16)x; }

// ---------------------------------------------------------------------------
// f32 -> bf16 weight conversion
// ---------------------------------------------------------------------------
__global__ void cvt_bf16_kernel(const float* __restrict__ src,
                                __bf16* __restrict__ dst, int n) {
  int i = blockIdx.x * blockDim.x + threadIdx.x;
  if (i < n) dst[i] = f2bf(src[i]);
}

// dec_w (256 x 16) -> decT bf16 (16 x 256)
__global__ void cvt_decT_kernel(const float* __restrict__ src,
                                __bf16* __restrict__ dst) {
  int i = blockIdx.x * blockDim.x + threadIdx.x;
  if (i < CH * 16) {
    int c = i >> 4, t = i & 15;
    dst[t * CH + c] = f2bf(src[i]);
  }
}

// ---------------------------------------------------------------------------
// Encoder: enc[pos][c] = sum_t ac[b, 8l+t]*acw[c][t] + bc[b, 8l+t]*bcw[c][t]
// One block per position, thread = channel; window staged in LDS.
// ---------------------------------------------------------------------------
__global__ __launch_bounds__(256)
void encoder_kernel(const float* __restrict__ ac, const float* __restrict__ bc,
                    const float* __restrict__ acw, const float* __restrict__ bcw,
                    float* __restrict__ enc) {
  int p = blockIdx.x;                 // global position
  int c = threadIdx.x;
  int b = p / LF, l = p % LF;
  __shared__ float win[32];
  if (c < 16)      win[c]      = ac[(size_t)b * SEQN + l * 8 + c];
  else if (c < 32) win[c]      = bc[(size_t)b * SEQN + l * 8 + (c - 16)];
  __syncthreads();
  float s = 0.f;
#pragma unroll
  for (int t = 0; t < 16; ++t)
    s += win[t] * acw[c * 16 + t] + win[16 + t] * bcw[c * 16 + t];
  enc[(size_t)p * CH + c] = s;
}

// ---------------------------------------------------------------------------
// LayerNorm over channels (wave32 per position, 8 channels/lane, shfl-xor
// reduction). Optional scalar PReLU applied before LN (block entry path).
// Input is either f32 (encoder output) or bf16 (residual stream).
// ---------------------------------------------------------------------------
__global__ __launch_bounds__(256)
void ln_kernel(const float* __restrict__ inF, const __bf16* __restrict__ inB,
               const float* __restrict__ gamma, const float* __restrict__ beta,
               const float* __restrict__ alpha,   // scalar PReLU slope or null
               __bf16* __restrict__ out, int mtot) {
  int wid = threadIdx.x >> 5, lane = threadIdx.x & 31;
  int p = blockIdx.x * 8 + wid;
  if (p >= mtot) return;
  float a0 = alpha ? alpha[0] : 0.f;
  float v[8], s = 0.f, ss = 0.f;
#pragma unroll
  for (int j = 0; j < 8; ++j) {
    int c = lane + 32 * j;
    float x = inF ? inF[(size_t)p * CH + c] : bf2f(inB[(size_t)p * CH + c]);
    if (alpha) x = (x >= 0.f) ? x : a0 * x;
    v[j] = x; s += x; ss += x * x;
  }
#pragma unroll
  for (int m = 16; m >= 1; m >>= 1) {
    s  += __shfl_xor(s,  m, 32);
    ss += __shfl_xor(ss, m, 32);
  }
  float mu  = s  * (1.f / CH);
  float var = ss * (1.f / CH) - mu * mu;
  float rs  = rsqrtf(var + EPSLN);
#pragma unroll
  for (int j = 0; j < 8; ++j) {
    int c = lane + 32 * j;
    out[(size_t)p * CH + c] = f2bf((v[j] - mu) * rs * gamma[c] + beta[c]);
  }
}

// ---------------------------------------------------------------------------
// WMMA GEMM: Y[m][n] = epilogue( sum_k A[m][k]*W[n][k] + bias[n] )
//   A: [M x K] bf16 channel-last activations (K contiguous)
//   W: [N x K] bf16 weight rows              (K contiguous)
// One wave per 16M x (16*NT)N strip; double-buffered fragment loads.
// EPI 0: bias, bf16 out          EPI 1: bias + residual add (bf16), bf16 out
// EPI 2: relu(bias+acc)*mulF32, bf16 out   EPI 3: raw f32 out, no bias
// ---------------------------------------------------------------------------
template <int EPI, int NT, int K>
__global__ __launch_bounds__(32)
void gemm16_kernel(const __bf16* __restrict__ A, const __bf16* __restrict__ W,
                   const float* __restrict__ bias,
                   __bf16* __restrict__ outB, float* __restrict__ outF,
                   const __bf16* __restrict__ addB, const float* __restrict__ mulF,
                   int M, int N) {
  int mt = blockIdx.x;
  int ntBase = blockIdx.y * NT;
  int lane  = threadIdx.x;
  int laneM = lane & 15;
  int hi    = lane >> 4;

  int aRow = mt * 16 + laneM; if (aRow >= M) aRow = M - 1;   // clamp partial tile
  int aKh  = hi * 8;     // A lanes 0-15: K 0-7 / 16-23 ; lanes 16-31: K 8-15 / 24-31
  int bKh  = hi * 8;     // B lanes 0-15: K 0-7 / 16-23 ; lanes 16-31: K 8-15 / 24-31

  const __bf16* Ap = A + (size_t)aRow * K + aKh;
  const __bf16* Wp[NT];
#pragma unroll
  for (int t = 0; t < NT; ++t)
    Wp[t] = W + (size_t)((ntBase + t) * 16 + laneM) * K + bKh;

  v8f acc[NT];
#pragma unroll
  for (int t = 0; t < NT; ++t) acc[t] = (v8f){};

  // prologue: load k-step 0 fragments
  v8bf ca0 = *(const v8bf*)(Ap);
  v8bf ca1 = *(const v8bf*)(Ap + 16);
  v8bf cb0[NT], cb1[NT];
#pragma unroll
  for (int t = 0; t < NT; ++t) {
    cb0[t] = *(const v8bf*)(Wp[t]);
    cb1[t] = *(const v8bf*)(Wp[t] + 16);
  }

#pragma unroll
  for (int kb = 0; kb < K; kb += 32) {
    // issue next step's loads before consuming the current fragments
    v8bf na0, na1, nb0[NT], nb1[NT];
    if (kb + 32 < K) {
      na0 = *(const v8bf*)(Ap + kb + 32);
      na1 = *(const v8bf*)(Ap + kb + 48);
#pragma unroll
      for (int t = 0; t < NT; ++t) {
        nb0[t] = *(const v8bf*)(Wp[t] + kb + 32);
        nb1[t] = *(const v8bf*)(Wp[t] + kb + 48);
      }
    }
    v16bf a;
#pragma unroll
    for (int i = 0; i < 8; ++i) { a[i] = ca0[i]; a[i + 8] = ca1[i]; }
#pragma unroll
    for (int t = 0; t < NT; ++t) {
      v16bf b;
#pragma unroll
      for (int i = 0; i < 8; ++i) { b[i] = cb0[t][i]; b[i + 8] = cb1[t][i]; }
      acc[t] = __builtin_amdgcn_wmma_f32_16x16x32_bf16(
          false, a, false, b, (short)0, acc[t], false, false);
    }
    ca0 = na0; ca1 = na1;
#pragma unroll
    for (int t = 0; t < NT; ++t) { cb0[t] = nb0[t]; cb1[t] = nb1[t]; }
  }

  int mBase = mt * 16 + (hi ? 8 : 0);
#pragma unroll
  for (int t = 0; t < NT; ++t) {
    int n = (ntBase + t) * 16 + laneM;
    float bia = (EPI == 3) ? 0.f : bias[n];
#pragma unroll
    for (int r = 0; r < 8; ++r) {
      int m = mBase + r;
      if (m >= M) continue;
      float v = acc[t][r];
      if (EPI == 0) {
        outB[(size_t)m * N + n] = f2bf(v + bia);
      } else if (EPI == 1) {
        v += bia + bf2f(addB[(size_t)m * N + n]);
        outB[(size_t)m * N + n] = f2bf(v);
      } else if (EPI == 2) {
        v += bia;
        v = (v > 0.f) ? v : 0.f;
        v *= mulF[(size_t)m * N + n];
        outB[(size_t)m * N + n] = f2bf(v);
      } else {
        outF[(size_t)m * N + n] = v;
      }
    }
  }
}

// ---------------------------------------------------------------------------
// Involution kernel generator: ker[p][k] = b2[k] + sum_c w2[k][c]*t[p][c]
// (t is the 64-wide bottleneck; w2 is 3x64 — tiny, thread-per-position)
// ---------------------------------------------------------------------------
__global__ __launch_bounds__(256)
void ker_kernel(const __bf16* __restrict__ t, const float* __restrict__ w2,
                const float* __restrict__ b2, float* __restrict__ ker, int mtot) {
  int p = blockIdx.x * blockDim.x + threadIdx.x;
  if (p >= mtot) return;
  float a0 = b2[0], a1 = b2[1], a2 = b2[2];
  for (int c = 0; c < 64; ++c) {
    float tv = bf2f(t[(size_t)p * 64 + c]);
    a0 += w2[c] * tv;
    a1 += w2[64 + c] * tv;
    a2 += w2[128 + c] * tv;
  }
  ker[p * 4 + 0] = a0; ker[p * 4 + 1] = a1; ker[p * 4 + 2] = a2;
}

// ---------------------------------------------------------------------------
// Involution apply + per-channel PReLU:
// out[l][c] = k0*m[l-1][c] + k1*m[l][c] + k2*m[l+1][c]  (zero pad per batch)
// ---------------------------------------------------------------------------
__global__ __launch_bounds__(256)
void inv_kernel(const __bf16* __restrict__ m, const float* __restrict__ ker,
                const float* __restrict__ alpha, __bf16* __restrict__ out) {
  int p = blockIdx.x, c = threadIdx.x;
  int l = p % LF;
  __shared__ float k3[3];
  if (c < 3) k3[c] = ker[p * 4 + c];
  __syncthreads();
  float v = k3[1] * bf2f(m[(size_t)p * CH + c]);
  if (l > 0)      v += k3[0] * bf2f(m[(size_t)(p - 1) * CH + c]);
  if (l < LF - 1) v += k3[2] * bf2f(m[(size_t)(p + 1) * CH + c]);
  float a = alpha[c];
  v = (v >= 0.f) ? v : a * v;
  out[(size_t)p * CH + c] = f2bf(v);
}

// ---------------------------------------------------------------------------
// Transposed-conv overlap-add: out[b][s] = P[l0][s%8] + P[l0-1][s%8+8]
// with l0 = s/8 (each sample overlapped by exactly 2 windows; stride 8, L 16)
// ---------------------------------------------------------------------------
__global__ __launch_bounds__(256)
void scatter_kernel(const float* __restrict__ P, float* __restrict__ out) {
  int i = blockIdx.x * blockDim.x + threadIdx.x;
  if (i >= 4 * SEQN) return;
  int b = i >> 15, s = i & (SEQN - 1);
  int l0 = s >> 3, r = s & 7;
  float o = 0.f;
  if (l0 < LF) o += P[((size_t)b * LF + l0) * 16 + r];
  if (l0 >= 1) o += P[((size_t)b * LF + l0 - 1) * 16 + r + 8];
  out[i] = o;
}

// ---------------------------------------------------------------------------
// Host orchestration
// ---------------------------------------------------------------------------
extern "C" void kernel_launch(void* const* d_in, const int* in_sizes, int n_in,
                              void* d_out, int out_size, void* d_ws, size_t ws_size,
                              hipStream_t stream) {
  const float* noisy_ac    = (const float*)d_in[0];
  const float* noisy_bc    = (const float*)d_in[1];
  const float* ac_w        = (const float*)d_in[2];
  const float* bc_w        = (const float*)d_in[3];
  const float* me_ln_g     = (const float*)d_in[4];
  const float* me_ln_b     = (const float*)d_in[5];
  const float* me_init_w   = (const float*)d_in[6];
  const float* me_init_b   = (const float*)d_in[7];
  const float* blk_ln_g    = (const float*)d_in[8];
  const float* blk_ln_b    = (const float*)d_in[9];
  const float* blk_act_a   = (const float*)d_in[10];
  const float* inv_w1      = (const float*)d_in[11];
  const float* inv_b1      = (const float*)d_in[12];
  const float* inv_w2      = (const float*)d_in[13];
  const float* inv_b2      = (const float*)d_in[14];
  const float* inv_prelu_a = (const float*)d_in[15];
  const float* skip_w      = (const float*)d_in[16];
  const float* skip_b      = (const float*)d_in[17];
  const float* me_final_w  = (const float*)d_in[18];
  const float* me_final_b  = (const float*)d_in[19];
  const float* dec_w       = (const float*)d_in[20];
  float* out = (float*)d_out;

  // --- workspace layout (256B aligned) ---
  char* ws = (char*)d_ws;
  size_t off = 0;
  auto alloc = [&](size_t bytes) -> char* {
    char* p = ws + off;
    off += (bytes + 255) & ~(size_t)255;
    return p;
  };
  float*  enc_f    = (float*) alloc((size_t)MT * CH * 4);
  __bf16* x_bf     = (__bf16*)alloc((size_t)MT * CH * 2);
  __bf16* xn_bf    = (__bf16*)alloc((size_t)MT * CH * 2);
  __bf16* m_bf     = (__bf16*)alloc((size_t)MT * CH * 2);
  __bf16* m2_bf    = (__bf16*)alloc((size_t)MT * CH * 2);
  __bf16* mask_bf  = (__bf16*)alloc((size_t)MT * CH * 2);
  __bf16* t64_bf   = (__bf16*)alloc((size_t)MT * 64 * 2);
  float*  ker_f    = (float*) alloc((size_t)MT * 4 * 4);
  float*  P_f      = (float*) alloc((size_t)MT * 16 * 4);
  __bf16* w_init   = (__bf16*)alloc(65536 * 2);
  __bf16* w_skip   = (__bf16*)alloc(4 * 65536 * 2);
  __bf16* w_inv1   = (__bf16*)alloc(4 * 3 * 64 * 256 * 2);
  __bf16* w_final  = (__bf16*)alloc(65536 * 2);
  __bf16* w_decT   = (__bf16*)alloc(16 * 256 * 2);

  // --- weight conversions f32 -> bf16 ---
  auto cvt = [&](const float* s, __bf16* d, int n) {
    cvt_bf16_kernel<<<(n + 255) / 256, 256, 0, stream>>>(s, d, n);
  };
  cvt(me_init_w,  w_init,  65536);
  cvt(skip_w,     w_skip,  4 * 65536);
  cvt(inv_w1,     w_inv1,  4 * 3 * 64 * 256);
  cvt(me_final_w, w_final, 65536);
  cvt_decT_kernel<<<(CH * 16 + 255) / 256, 256, 0, stream>>>(dec_w, w_decT);

  const int mtiles  = (MT + 15) / 16;   // 1024
  const dim3 g256(mtiles, CH / 64);     // NT=4 strips of 64 N
  const dim3 g64 (mtiles, 1);           // NT=4 covers all 64 N
  const dim3 g16 (mtiles, 1);           // NT=1, N=16
  const int lnGrid = (MT + 7) / 8;

  // --- encoder + initial LN + bottleneck matmul ---
  encoder_kernel<<<MT, 256, 0, stream>>>(noisy_ac, noisy_bc, ac_w, bc_w, enc_f);
  ln_kernel<<<lnGrid, 256, 0, stream>>>(enc_f, nullptr, me_ln_g, me_ln_b,
                                        nullptr, xn_bf, MT);
  gemm16_kernel<0, 4, CH><<<g256, 32, 0, stream>>>(xn_bf, w_init, me_init_b,
                                                   x_bf, nullptr, nullptr, nullptr,
                                                   MT, CH);

  // --- 4 blocks x 3 involutions ---
  for (int blk = 0; blk < 4; ++blk) {
    ln_kernel<<<lnGrid, 256, 0, stream>>>(nullptr, x_bf,
                                          blk_ln_g + blk * CH, blk_ln_b + blk * CH,
                                          blk_act_a + blk, xn_bf, MT);
    const __bf16* mcur = xn_bf;
    for (int p = 0; p < 3; ++p) {
      int bp = blk * 3 + p;
      gemm16_kernel<0, 4, CH><<<g64, 32, 0, stream>>>(
          mcur, w_inv1 + (size_t)bp * 64 * 256, inv_b1 + bp * 64,
          t64_bf, nullptr, nullptr, nullptr, MT, 64);
      ker_kernel<<<(MT + 255) / 256, 256, 0, stream>>>(t64_bf, inv_w2 + bp * 192,
                                                       inv_b2 + bp * 3, ker_f, MT);
      __bf16* mout = (p & 1) ? m2_bf : m_bf;
      inv_kernel<<<MT, 256, 0, stream>>>(mcur, ker_f, inv_prelu_a + bp * CH, mout);
      mcur = mout;
    }
    // x = m + skip_w @ xn + skip_b   (mcur == m_bf after 3 iterations)
    gemm16_kernel<1, 4, CH><<<g256, 32, 0, stream>>>(
        xn_bf, w_skip + (size_t)blk * 65536, skip_b + blk * CH,
        x_bf, nullptr, mcur, nullptr, MT, CH);
  }

  // --- mask = relu(me_final @ x + b), fused masked = enc * mask ---
  gemm16_kernel<2, 4, CH><<<g256, 32, 0, stream>>>(x_bf, w_final, me_final_b,
                                                   mask_bf, nullptr, nullptr, enc_f,
                                                   MT, CH);

  // --- deconv projection P = masked @ dec_w  (Lf x 16), then overlap-add ---
  gemm16_kernel<3, 1, CH><<<g16, 32, 0, stream>>>(mask_bf, w_decT, nullptr,
                                                  nullptr, P_f, nullptr, nullptr,
                                                  MT, 16);
  scatter_kernel<<<(4 * SEQN + 255) / 256, 256, 0, stream>>>(P_f, out);
}
